// GCN_46995532153066
// MI455X (gfx1250) — matrix operand order repaired and checked by reading, not compiled, not run on hardware
//
#include <hip/hip_runtime.h>
#include <hip/hip_bf16.h>
#include <math.h>

// ---------------------------------------------------------------------------
// GCN + SAGE + MLP pipeline for MI455X (gfx1250, wave32).
// Dense GEMMs via v_wmma_f32_16x16x32_f16; weights staged into LDS in
// WMMA-B fragment order (each fragment = 2x ds_load_b128 per lane).
// Sparse phases via L2-resident f32 atomic scatter-add (node tables are
// 12.8 MB; the whole graph state lives in the 192 MB L2).
// NOTE: row count is assumed to be a multiple of 16 (N_NODES = 50000 = 3125*16);
// the only tile guard is the block-uniform `row0 >= nrows` break.
// ---------------------------------------------------------------------------

typedef _Float16 half_t;
typedef __attribute__((ext_vector_type(16))) _Float16 v16h;
typedef __attribute__((ext_vector_type(8)))  float    v8f;

#define EPS_BN 1e-5f
#define ROWTILES 8   // 16-row tiles per block (weights staged once per block)

__device__ __forceinline__ v8f wmma_f16(v16h a, v16h b, v8f c) {
  return __builtin_amdgcn_wmma_f32_16x16x32_f16(false, a, false, b, (short)0, c,
                                                false, false);
}

// A-fragment: 16x32 f16 tile from row-major LDS (lda halves per row).
// lane L: m = L&15, hl = L>>4; contiguous 8-half runs -> 2x ds_load_b128.
__device__ __forceinline__ v16h load_a_frag(const half_t* As, int lda, int k0, int lane) {
  const int m = lane & 15, hl = lane >> 4;
  v16h a;
#pragma unroll
  for (int v = 0; v < 8; ++v) {
    const int g = v >> 2, vv = v & 3;
    const int k = k0 + g * 16 + hl * 8 + 2 * vv;
    a[2 * v]     = As[m * lda + k];
    a[2 * v + 1] = As[m * lda + k + 1];
  }
  return a;
}

// B-fragment from pre-swizzled LDS: 16 contiguous halves per lane (32B aligned).
__device__ __forceinline__ v16h load_b_swz(const half_t* Bs, int frag, int lane) {
  return *(const v16h*)(Bs + frag * 512 + lane * 16);
}

// Stage a K=64 x NW (64 or 32) f32 row-major weight into WMMA-B fragment order.
// dest[frag*512 + lane*16 + e] = W[k][n], frag = (n>>4)*2 + (k>>5),
// lane = ((k>>3)&1)*16 + (n&15), e = (k&7) + ((k&31)>>4)*8.
template <int NW>
__device__ __forceinline__ void stage_b_swz(const float* __restrict__ W,
                                            half_t* __restrict__ Bs, int t) {
  for (int i = t; i < 64 * NW; i += 128) {
    const int k = i / NW, n = i % NW;           // NW power of two -> shifts
    const int frag = (n >> 4) * 2 + (k >> 5);
    const int kk = k & 31;
    const int d = frag * 512 + ((((kk >> 3) & 1) << 4) + (n & 15)) * 16 + ((kk & 7) + ((kk >> 4) << 3));
    Bs[d] = (half_t)W[i];
  }
}

// ---------------------------------------------------------------------------
__global__ __launch_bounds__(256) void k_degree(const int* __restrict__ dst,
                                                float* __restrict__ cnt, int E) {
  int e = blockIdx.x * 256 + threadIdx.x;
  if (e < E) unsafeAtomicAdd(&cnt[dst[e]], 1.0f);
}

__global__ __launch_bounds__(256) void k_dinv(const float* __restrict__ cnt,
                                              float* __restrict__ dinv, int N) {
  int v = blockIdx.x * 256 + threadIdx.x;
  if (v < N) dinv[v] = rsqrtf(cnt[v] + 1.0f);
}

// ---------------------------------------------------------------------------
// h0 = x @ W_gcn. Block = 128 thr / 4 waves; W staged once, ROWTILES tiles.
// ---------------------------------------------------------------------------
__global__ __launch_bounds__(128) void k_gemm_x(const float* __restrict__ A,
                                                const float* __restrict__ W,
                                                float* __restrict__ out, int nrows) {
  __shared__ alignas(32) half_t Bs[8 * 512];   // 4 n-tiles x 2 k-halves
  __shared__ alignas(32) half_t As[16 * 64];
  const int t = threadIdx.x;
  const int wave = t >> 5, lane = t & 31;
  const int hl = lane >> 4, n = lane & 15;

  stage_b_swz<64>(W, Bs, t);

  const int tile0 = blockIdx.x * ROWTILES;
  for (int ti = 0; ti < ROWTILES; ++ti) {
    const int row0 = (tile0 + ti) * 16;
    if (row0 >= nrows) break;                  // block-uniform tile guard
    __syncthreads();                           // W staged / previous tile consumed
    for (int i = t; i < 16 * 64; i += 128)
      As[i] = (half_t)A[row0 * 64 + i];        // 16 full rows, coalesced
    __syncthreads();

    v8f c = {};
    c = wmma_f16(load_a_frag(As, 64, 0, lane),  load_b_swz(Bs, wave * 2 + 0, lane), c);
    c = wmma_f16(load_a_frag(As, 64, 32, lane), load_b_swz(Bs, wave * 2 + 1, lane), c);

#pragma unroll
    for (int r = 0; r < 8; ++r)
      out[(row0 + r + 8 * hl) * 64 + wave * 16 + n] = c[r];
  }
}

// ---------------------------------------------------------------------------
// GCN edge scatter: agg[dst] += h0[src] * dinv[src]*dinv[dst]
// ---------------------------------------------------------------------------
__global__ __launch_bounds__(256) void k_gcn_scatter(const float* __restrict__ h0,
                                                     const int* __restrict__ src,
                                                     const int* __restrict__ dst,
                                                     const float* __restrict__ dinv,
                                                     float* __restrict__ agg, int total) {
  int tid = blockIdx.x * 256 + threadIdx.x;
  if (tid >= total) return;
  int e = tid >> 4, c4 = (tid & 15) << 2;
  int s = src[e], d = dst[e];
  float nrm = dinv[s] * dinv[d];
  float4 v = *(const float4*)(h0 + s * 64 + c4);
  float* o = agg + d * 64 + c4;
  unsafeAtomicAdd(o + 0, v.x * nrm);
  unsafeAtomicAdd(o + 1, v.y * nrm);
  unsafeAtomicAdd(o + 2, v.z * nrm);
  unsafeAtomicAdd(o + 3, v.w * nrm);
}

__global__ __launch_bounds__(256) void k_gcn_finish(
    const float* __restrict__ h0, const float* __restrict__ agg,
    const float* __restrict__ dinv, const float* __restrict__ b_gcn,
    const float* __restrict__ g1, const float* __restrict__ be1,
    const float* __restrict__ m1, const float* __restrict__ v1,
    float* __restrict__ h1, int N) {
  int i = blockIdx.x * 256 + threadIdx.x;
  if (i >= N * 64) return;
  int nd = i >> 6, dch = i & 63;
  float di = dinv[nd];
  float val = agg[i] + h0[i] * di * di + b_gcn[dch];
  val = (val - m1[dch]) * (g1[dch] * rsqrtf(v1[dch] + EPS_BN)) + be1[dch];
  h1[i] = fmaxf(val, 0.0f);
}

__global__ __launch_bounds__(256) void k_sage_scatter(const float* __restrict__ h1,
                                                      const int* __restrict__ src,
                                                      const int* __restrict__ dst,
                                                      float* __restrict__ nb, int total) {
  int tid = blockIdx.x * 256 + threadIdx.x;
  if (tid >= total) return;
  int e = tid >> 4, c4 = (tid & 15) << 2;
  int s = src[e], d = dst[e];
  float4 v = *(const float4*)(h1 + s * 64 + c4);
  float* o = nb + d * 64 + c4;
  unsafeAtomicAdd(o + 0, v.x);
  unsafeAtomicAdd(o + 1, v.y);
  unsafeAtomicAdd(o + 2, v.z);
  unsafeAtomicAdd(o + 3, v.w);
}

// ---------------------------------------------------------------------------
// Fused tail: h2 = relu(bn2(mean(nb)@Wl + h1@Wr + b)); t = relu(h2@Wa + ba);
//             logits = t@Wb + bb; log_softmax. Weights staged once per block.
// ---------------------------------------------------------------------------
__global__ __launch_bounds__(128) void k_sage_mlp(
    const float* __restrict__ h1, const float* __restrict__ nb,
    const float* __restrict__ cnt,
    const float* __restrict__ Wl, const float* __restrict__ Wr,
    const float* __restrict__ b_sage,
    const float* __restrict__ g2, const float* __restrict__ be2,
    const float* __restrict__ m2, const float* __restrict__ v2,
    const float* __restrict__ Wa, const float* __restrict__ ba,
    const float* __restrict__ Wb, const float* __restrict__ bb,
    float* __restrict__ out, int nrows) {
  __shared__ alignas(32) half_t Bl[8 * 512];   // Wl frags
  __shared__ alignas(32) half_t Br[8 * 512];   // Wr frags
  __shared__ alignas(32) half_t Ba[4 * 512];   // Wa frags
  __shared__ alignas(32) half_t Am[16 * 64];   // mean-neighbor tile
  __shared__ alignas(32) half_t Ah[16 * 64];   // h1 tile
  __shared__ alignas(32) half_t H2[16 * 64];   // post-BN2 hidden
  __shared__ float  T[16 * 32];
  __shared__ float  Lg[16][12];

  const int t = threadIdx.x;
  const int wave = t >> 5, lane = t & 31;
  const int hl = lane >> 4, n = lane & 15;

  stage_b_swz<64>(Wl, Bl, t);
  stage_b_swz<64>(Wr, Br, t);
  stage_b_swz<32>(Wa, Ba, t);

  const int tile0 = blockIdx.x * ROWTILES;
  for (int ti = 0; ti < ROWTILES; ++ti) {
    const int row0 = (tile0 + ti) * 16;
    if (row0 >= nrows) break;                  // block-uniform tile guard
    __syncthreads();                           // weights staged / prev tile done
    for (int i = t; i < 16 * 64; i += 128) {
      int rr = row0 + (i >> 6);
      float inv = __builtin_amdgcn_rcpf(fmaxf(cnt[rr], 1.0f));  // v_rcp_f32
      Am[i] = (half_t)(nb[row0 * 64 + i] * inv);
      Ah[i] = (half_t)h1[row0 * 64 + i];
    }
    __syncthreads();

    // --- SAGE GEMMs: mean@Wl + h1@Wr in one f32 accumulator tile ---
    v8f c = {};
    c = wmma_f16(load_a_frag(Am, 64, 0, lane),  load_b_swz(Bl, wave * 2 + 0, lane), c);
    c = wmma_f16(load_a_frag(Am, 64, 32, lane), load_b_swz(Bl, wave * 2 + 1, lane), c);
    c = wmma_f16(load_a_frag(Ah, 64, 0, lane),  load_b_swz(Br, wave * 2 + 0, lane), c);
    c = wmma_f16(load_a_frag(Ah, 64, 32, lane), load_b_swz(Br, wave * 2 + 1, lane), c);

    {
      const int col = wave * 16 + n;
      const float sc = g2[col] * rsqrtf(v2[col] + EPS_BN);
      const float mu = m2[col], bt = be2[col], bs = b_sage[col];
#pragma unroll
      for (int r = 0; r < 8; ++r) {
        float val = (c[r] + bs - mu) * sc + bt;
        H2[(r + 8 * hl) * 64 + col] = (half_t)fmaxf(val, 0.0f);
      }
    }
    __syncthreads();

    // --- MLP layer A (64 -> 32): waves 0,1 own the two 16-col tiles ---
    if (wave < 2) {
      v8f c2 = {};
      c2 = wmma_f16(load_a_frag(H2, 64, 0, lane),  load_b_swz(Ba, wave * 2 + 0, lane), c2);
      c2 = wmma_f16(load_a_frag(H2, 64, 32, lane), load_b_swz(Ba, wave * 2 + 1, lane), c2);
      const int col = wave * 16 + n;
      const float bav = ba[col];
#pragma unroll
      for (int r = 0; r < 8; ++r)
        T[(r + 8 * hl) * 32 + col] = fmaxf(c2[r] + bav, 0.0f);
    }
    __syncthreads();

    // --- MLP layer B (32 -> 10), scalar ---
    if (t < 160) {
      int r = t / 10, cc = t - 10 * r;
      float s = bb[cc];
#pragma unroll
      for (int j = 0; j < 32; ++j) s += T[r * 32 + j] * Wb[j * 10 + cc];
      Lg[r][cc] = s;
    }
    __syncthreads();

    // --- log_softmax per row ---
    if (t < 16) {
      float mx = Lg[t][0];
#pragma unroll
      for (int cj = 1; cj < 10; ++cj) mx = fmaxf(mx, Lg[t][cj]);
      float s = 0.0f;
#pragma unroll
      for (int cj = 0; cj < 10; ++cj) s += __expf(Lg[t][cj] - mx);
      float ls = __logf(s);
#pragma unroll
      for (int cj = 0; cj < 10; ++cj)
        out[(row0 + t) * 10 + cj] = Lg[t][cj] - mx - ls;
    }
    __syncthreads();                           // Lg/T consumed before next tile
  }
}

// ---------------------------------------------------------------------------
extern "C" void kernel_launch(void* const* d_in, const int* in_sizes, int n_in,
                              void* d_out, int out_size, void* d_ws, size_t ws_size,
                              hipStream_t stream) {
  const float* x     = (const float*)d_in[0];
  const int*   ei    = (const int*)  d_in[1];
  const float* Wgcn  = (const float*)d_in[2];
  const float* bgcn  = (const float*)d_in[3];
  const float* g1    = (const float*)d_in[4];
  const float* be1   = (const float*)d_in[5];
  const float* m1    = (const float*)d_in[6];
  const float* v1    = (const float*)d_in[7];
  const float* Wl    = (const float*)d_in[8];
  const float* Wr    = (const float*)d_in[9];
  const float* bsage = (const float*)d_in[10];
  const float* g2    = (const float*)d_in[11];
  const float* be2   = (const float*)d_in[12];
  const float* m2    = (const float*)d_in[13];
  const float* v2    = (const float*)d_in[14];
  const float* Wa    = (const float*)d_in[15];
  const float* ba    = (const float*)d_in[16];
  const float* Wb    = (const float*)d_in[17];
  const float* bb    = (const float*)d_in[18];
  float* out = (float*)d_out;

  const int N = in_sizes[0] / 64;
  const int E = in_sizes[1] / 2;
  const int* src = ei;
  const int* dst = ei + E;

  // ws layout (floats): [cnt N][agg 64N][nb 64N] (zeroed) [dinv N][h0 64N][h1 64N]
  float* ws   = (float*)d_ws;
  float* cnt  = ws;
  float* agg  = cnt + N;
  float* nb   = agg + (size_t)64 * N;
  float* dinv = nb  + (size_t)64 * N;
  float* h0   = dinv + N;
  float* h1   = h0  + (size_t)64 * N;

  hipMemsetAsync(ws, 0, (size_t)(129 * (size_t)N) * sizeof(float), stream);

  const int nTiles = (N + 15) / 16;            // N % 16 == 0 for this problem
  const int nBlocks = (nTiles + ROWTILES - 1) / ROWTILES;
  const int eTot = E * 16;

  k_degree<<<(E + 255) / 256, 256, 0, stream>>>(dst, cnt, E);
  k_dinv<<<(N + 255) / 256, 256, 0, stream>>>(cnt, dinv, N);
  k_gemm_x<<<nBlocks, 128, 0, stream>>>(x, Wgcn, h0, N);
  k_gcn_scatter<<<(eTot + 255) / 256, 256, 0, stream>>>(h0, src, dst, dinv, agg, eTot);
  k_gcn_finish<<<(N * 64 + 255) / 256, 256, 0, stream>>>(h0, agg, dinv, bgcn, g1, be1,
                                                         m1, v1, h1, N);
  k_sage_scatter<<<(eTot + 255) / 256, 256, 0, stream>>>(h1, src, dst, nb, eTot);
  k_sage_mlp<<<nBlocks, 128, 0, stream>>>(h1, nb, cnt, Wl, Wr, bsage, g2, be2, m2, v2,
                                          Wa, ba, Wb, bb, out, N);
}
